// Metric_PR_Base_42863773614715
// MI455X (gfx1250) — compile-verified, matrix operand order checked by reading, not run on hardware
//
#include <hip/hip_runtime.h>
#include <hip/hip_bf16.h>

typedef __attribute__((ext_vector_type(16))) _Float16 v16h;
typedef __attribute__((ext_vector_type(8)))  float    v8f;

#define PR_EPS 1e-6f
#define NCLASS 16

// d_ws layout (float): [0..15] intersection, [16..31] gt_sum, [32..47] pred_sum
__global__ __launch_bounds__(64) void pr_zero_ws(float* __restrict__ ws) {
    if (threadIdx.x < 3 * NCLASS) ws[threadIdx.x] = 0.0f;
}

// Load one 32x16 block's slice for this lane: class n, 16 samples, stride 64B.
// Across the wave each load instruction covers two full contiguous 64B rows.
// Non-temporal: 512MB streamed once > 192MB L2, don't pollute it.
__device__ __forceinline__ void pr_loadblk(const float* __restrict__ p,
                                           const float* __restrict__ g,
                                           long long base,
                                           float (&pf)[16], float (&gf)[16]) {
#pragma unroll
    for (int h = 0; h < 16; ++h) {
        pf[h] = __builtin_nontemporal_load(p + base + (long long)h * NCLASS);
        gf[h] = __builtin_nontemporal_load(g + base + (long long)h * NCLASS);
    }
}

// D = ones(16x32) x B(32x16) + C : every row of D = per-class column sums of B.
__device__ __forceinline__ void pr_accum(const float (&pf)[16], const float (&gf)[16],
                                         const v16h& ones,
                                         v8f& accP, v8f& accG, v8f& accI) {
    v16h pb, gb;
#pragma unroll
    for (int h = 0; h < 16; ++h) {
        pb[h] = (_Float16)pf[h];   // exact for {0,1}
        gb[h] = (_Float16)gf[h];
    }
    v16h ib = pb * gb;             // v_pk_mul_f16, exact for indicators
    accP = __builtin_amdgcn_wmma_f32_16x16x32_f16(false, ones, false, pb,
                                                  (short)0, accP, false, false);
    accG = __builtin_amdgcn_wmma_f32_16x16x32_f16(false, ones, false, gb,
                                                  (short)0, accG, false, false);
    accI = __builtin_amdgcn_wmma_f32_16x16x32_f16(false, ones, false, ib,
                                                  (short)0, accI, false, false);
}

__global__ __launch_bounds__(256) void pr_reduce_wmma(const float* __restrict__ pred,
                                                      const float* __restrict__ gt,
                                                      float* __restrict__ ws,
                                                      int nBlocks) {
    const int lane = threadIdx.x & 31;
    // wave index is uniform within a wave: pin it to an SGPR so all trip-count
    // control flow is scalar branches (EXEC provably all-ones at every WMMA).
    const int wave = __builtin_amdgcn_readfirstlane(
        (int)(blockIdx.x * (blockDim.x >> 5)) + (threadIdx.x >> 5));
    const int nWaves = (int)(gridDim.x * (blockDim.x >> 5));

    const int n  = lane & 15;         // class column this lane feeds into B
    const int k0 = (lane >> 4) << 4;  // sample sub-range (0 or 16)

    v16h ones;
#pragma unroll
    for (int i = 0; i < 16; ++i) ones[i] = (_Float16)1.0f;

    v8f accI = {};  // intersection
    v8f accG = {};  // gt_sum
    v8f accP = {};  // pred_sum

    // number of 32-sample blocks this wave owns (scalar)
    const int m = (wave < nBlocks) ? ((nBlocks - 1 - wave) / nWaves + 1) : 0;

    long long base = (long long)wave * (32 * NCLASS) + (long long)k0 * NCLASS + n;
    const long long stride = (long long)nWaves * (32 * NCLASS);

    if (m > 0) {
        float pf0[16], gf0[16], pf1[16], gf1[16];
        pr_loadblk(pred, gt, base, pf0, gf0);
        int i = 0;
        // Software-pipelined ping-pong: issue next block's 32 loads, then do
        // the 3 WMMAs on the resident block; one loadcnt wait per block.
        while (true) {
            if (i + 1 < m) pr_loadblk(pred, gt, base + stride, pf1, gf1);
            pr_accum(pf0, gf0, ones, accP, accG, accI);
            ++i; base += stride;
            if (i >= m) break;

            if (i + 1 < m) pr_loadblk(pred, gt, base + stride, pf0, gf0);
            pr_accum(pf1, gf1, ones, accP, accG, accI);
            ++i; base += stride;
            if (i >= m) break;
        }
    }

    // C/D layout: VGPR0, lanes 0..15 = D[0][lane] -> class 'lane' partial sum.
    if (lane < 16) {
        atomicAdd(&ws[lane],              accI[0]);
        atomicAdd(&ws[NCLASS + lane],     accG[0]);
        atomicAdd(&ws[2 * NCLASS + lane], accP[0]);
    }
}

// out: [0..15] precisions, [16..31] recalls, [32..47] intersection,
//      [48..63] gt_sum, [64..79] pred_sum
__global__ __launch_bounds__(32) void pr_finalize(const float* __restrict__ ws,
                                                  float* __restrict__ out) {
    const int i = threadIdx.x;
    if (i < NCLASS) {
        const float inter = ws[i];
        const float gts   = ws[NCLASS + i];
        const float prs   = ws[2 * NCLASS + i];
        out[i]              = (inter + PR_EPS) / (prs + PR_EPS);  // precision
        out[NCLASS + i]     = (inter + PR_EPS) / (gts + PR_EPS);  // recall
        out[2 * NCLASS + i] = inter;
        out[3 * NCLASS + i] = gts;
        out[4 * NCLASS + i] = prs;
    }
}

extern "C" void kernel_launch(void* const* d_in, const int* in_sizes, int n_in,
                              void* d_out, int out_size, void* d_ws, size_t ws_size,
                              hipStream_t stream) {
    const float* pred = (const float*)d_in[0];
    const float* gt   = (const float*)d_in[1];
    float* out = (float*)d_out;
    float* ws  = (float*)d_ws;

    const int n       = in_sizes[0] / NCLASS;  // 4,194,304 rows
    const int nBlocks = n / 32;                // 32 samples per WMMA block (N % 32 == 0)

    pr_zero_ws<<<1, 64, 0, stream>>>(ws);

    // 2048 WGs x 256 threads = 16,384 waves -> 8 blocks (32 KB traffic) per wave
    pr_reduce_wmma<<<2048, 256, 0, stream>>>(pred, gt, ws, nBlocks);

    pr_finalize<<<1, 32, 0, stream>>>(ws, out);
}
